// LlamaMLP_13469017441058
// MI455X (gfx1250) — compile-verified, matrix operand order checked by reading, not run on hardware
//
#include <hip/hip_runtime.h>

#define H_DIM 1024
#define I_DIM 2816
#define E_R   7
#define NTOK  8192          // B*S = 4*2048
#define TM    16            // tokens per workgroup (one WMMA M tile)
#define IC    128           // I-chunk per iteration
#define XSTR  (H_DIM + 8)   // padded LDS row stride (bf16 elems) -> conflict-free ds_load
#define ASTR  (IC + 8)

typedef __bf16 bf16_t;
typedef __attribute__((ext_vector_type(8)))  __bf16 v8bf;
typedef __attribute__((ext_vector_type(16))) __bf16 v16bf;
typedef __attribute__((ext_vector_type(8)))  float  v8f;

// ---- WMMA helpers -------------------------------------------------------

static __device__ __forceinline__ v16bf ld_frag(const bf16_t* q) {
  // 16 bf16 for one lane of an A/B fragment: K k0..k0+7 and k0+16..k0+23
  v8bf lo = *(const v8bf*)q;
  v8bf hi = *(const v8bf*)(q + 16);
  v16bf r;
#pragma unroll
  for (int i = 0; i < 8; ++i) { r[i] = lo[i]; r[i + 8] = hi[i]; }
  return r;
}

static __device__ __forceinline__ v16bf ld_a(const bf16_t* base, int kbase,
                                             int stride, int lane) {
  // A-matrix 16x32 bf16 layout: lanes 0-15 -> M=lane, K 0..7 & 16..23;
  // lanes 16-31 -> M=lane-16, K 8..15 & 24..31.
  const bf16_t* q = base + (lane & 15) * stride + kbase + ((lane >> 4) << 3);
  return ld_frag(q);
}

static __device__ __forceinline__ v8f wmma_bf16(v16bf a, v16bf b, v8f c) {
  return __builtin_amdgcn_wmma_f32_16x16x32_bf16(
      /*neg_a=*/false, a, /*neg_b=*/false, b,
      /*c_mod=*/(short)0, c, /*reuse_a=*/false, /*reuse_b=*/false);
}

// Fast SiLU: x * rcp(1+exp(-x)) — single v_rcp_f32, plenty for bf16 WMMA math.
static __device__ __forceinline__ float fast_silu(float x) {
  return x * __builtin_amdgcn_rcpf(1.0f + __expf(-x));
}

// ---- f32 -> bf16 conversion (makes the 138MB weight set L2-resident) ----

__global__ void cvt_kernel(const float* __restrict__ in, bf16_t* __restrict__ out,
                           int n) {
  int idx = (blockIdx.x * blockDim.x + threadIdx.x) * 4;
  if (idx + 3 < n) {
    float4 v = *(const float4*)(in + idx);
    out[idx + 0] = (bf16_t)v.x;
    out[idx + 1] = (bf16_t)v.y;
    out[idx + 2] = (bf16_t)v.z;
    out[idx + 3] = (bf16_t)v.w;
  } else {
    for (int k = idx; k < n; ++k) out[k] = (bf16_t)in[k];
  }
}

// ---- Router: logits -> softmax -> top-2 -> renormalized weights ---------

__global__ void router_kernel(const float* __restrict__ x, const float* __restrict__ wr,
                              const float* __restrict__ rb,
                              int* __restrict__ i0, int* __restrict__ i1,
                              float* __restrict__ w0, float* __restrict__ w1) {
  int t = blockIdx.x * blockDim.x + threadIdx.x;
  if (t >= NTOK) return;
  const float4* xr  = (const float4*)(x + (size_t)t * H_DIM);
  const float4* wrv = (const float4*)wr;
  float acc[E_R];
#pragma unroll
  for (int e = 0; e < E_R; ++e) acc[e] = 0.0f;
  for (int k = 0; k < H_DIM / 4; ++k) {
    float4 xv = xr[k];
#pragma unroll
    for (int e = 0; e < E_R; ++e) {
      float4 wv = wrv[e * (H_DIM / 4) + k];
      acc[e] += xv.x * wv.x + xv.y * wv.y + xv.z * wv.z + xv.w * wv.w;
    }
  }
  float p[E_R];
  float m = -3.0e38f;
#pragma unroll
  for (int e = 0; e < E_R; ++e) { p[e] = acc[e] + rb[e]; m = fmaxf(m, p[e]); }
#pragma unroll
  for (int e = 0; e < E_R; ++e) p[e] = __expf(p[e] - m);
  // softmax denominator cancels in the top-2 renormalization
  int a0 = 0; float b0 = p[0];
#pragma unroll
  for (int e = 1; e < E_R; ++e) if (p[e] > b0) { b0 = p[e]; a0 = e; }
  int a1 = (a0 == 0) ? 1 : 0; float b1 = p[a1];
#pragma unroll
  for (int e = 0; e < E_R; ++e) if (e != a0 && p[e] > b1) { b1 = p[e]; a1 = e; }
  float inv = 1.0f / (b0 + b1);
  i0[t] = a0; i1[t] = a1; w0[t] = b0 * inv; w1[t] = b1 * inv;
}

// ---- Deterministic per-expert token gather lists ------------------------

__global__ void build_lists_kernel(const int* __restrict__ i0, const int* __restrict__ i1,
                                   const float* __restrict__ w0, const float* __restrict__ w1,
                                   int* __restrict__ ltok, float* __restrict__ lw,
                                   int* __restrict__ counts) {
  int e = threadIdx.x;
  if (e >= E_R) return;
  int c = 0;
  for (int t = 0; t < NTOK; ++t) {
    if (i0[t] == e) { ltok[e * NTOK + c] = t; lw[e * NTOK + c] = w0[t]; ++c; }
    if (i1[t] == e) { ltok[e * NTOK + c] = t; lw[e * NTOK + c] = w1[t]; ++c; }
  }
  counts[e] = c;
}

// ---- Fused SwiGLU expert: out[+]= coef * down(silu(x g^T) * (x u^T)) ----
// One block = 16 (gathered) tokens. 8 waves.
// Phase A: wave w computes gate & up 16x16 tiles for act cols [w*16,w*16+16),
//          K = H via 32 bf16-WMMAs each, SiLU*up combined in-registers.
// Phase B: wave w accumulates out cols [w*128, w*128+128) (8 C tiles, 64 VGPRs),
//          K = IC chunk from LDS activations, weights streamed from L2 (bf16).

__global__ __launch_bounds__(256, 1) void moe_ffn_kernel(
    const bf16_t* __restrict__ xb,
    const bf16_t* __restrict__ gw, const bf16_t* __restrict__ uw,
    const bf16_t* __restrict__ dw,
    const int* __restrict__ ltok, const float* __restrict__ lw,
    const int* __restrict__ counts,
    float* __restrict__ out, int expert) {
  __shared__ __align__(16) bf16_t Xs[TM * XSTR];
  __shared__ __align__(16) bf16_t As[TM * ASTR];
  __shared__ int   tok_s[TM];
  __shared__ float coef_s[TM];

  const int tile = blockIdx.x;
  const bf16_t* gp = gw; const bf16_t* up_ = uw; const bf16_t* dp = dw;
  const int* lt = ltok; const float* lww = lw;
  int count = NTOK;
  if (expert >= 0) {
    size_t off = (size_t)expert * I_DIM * H_DIM;
    gp += off; up_ += off; dp += off;
    lt  += (size_t)expert * NTOK;
    lww += (size_t)expert * NTOK;
    count = counts[expert];
    if (tile * TM >= count) return;   // uniform early-exit (EXEC stays all-1s)
  }

  const int tid = threadIdx.x;
  if (tid < TM) {
    int g = tile * TM + tid;
    if (expert >= 0) {
      int gi = (g < count) ? g : (count - 1);
      tok_s[tid]  = lt[gi];
      coef_s[tid] = (g < count) ? lww[gi] : 0.0f;   // pad rows contribute 0
    } else {
      tok_s[tid]  = g;
      coef_s[tid] = 1.0f;
    }
  }
  __syncthreads();

  {  // stage X tile (gathered rows) into padded LDS: 16 x 1024 bf16
    uint4* dst = (uint4*)Xs;
    for (int i = tid; i < TM * (H_DIM / 8); i += 256) {
      int r = i >> 7;                 // 128 uint4 of data per row
      int c = i & 127;
      dst[r * (XSTR / 8) + c] = ((const uint4*)(xb + (size_t)tok_s[r] * H_DIM))[c];
    }
  }
  __syncthreads();

  const int wv    = tid >> 5;
  const int lane  = tid & 31;
  const int lhalf = (lane >> 4) << 3;   // 0 or 8
  const int l15   = lane & 15;

  v8f acc[8];
#pragma unroll
  for (int nt = 0; nt < 8; ++nt)
#pragma unroll
    for (int j = 0; j < 8; ++j) acc[nt][j] = 0.0f;

  for (int ic = 0; ic < I_DIM; ic += IC) {
    // -------- Phase A: gate & up, K = H --------
    v8f cg, cu;
#pragma unroll
    for (int j = 0; j < 8; ++j) { cg[j] = 0.0f; cu[j] = 0.0f; }

    const int nrow = ic + wv * 16 + l15;   // row of g/u weight (I dimension)
    const bf16_t* gRow = gp  + (size_t)nrow * H_DIM + lhalf;
    const bf16_t* uRow = up_ + (size_t)nrow * H_DIM + lhalf;
    if (nrow + IC < I_DIM) {               // prefetch next chunk's weight rows
      __builtin_prefetch(gp  + (size_t)(nrow + IC) * H_DIM, 0, 0);
      __builtin_prefetch(up_ + (size_t)(nrow + IC) * H_DIM, 0, 0);
    }
#pragma unroll 4
    for (int kk = 0; kk < H_DIM; kk += 32) {
      v16bf a  = ld_a(Xs, kk, XSTR, lane);
      v16bf bg = ld_frag(gRow + kk);
      v16bf bu = ld_frag(uRow + kk);
      cg = wmma_bf16(a, bg, cg);
      cu = wmma_bf16(a, bu, cu);
    }

    __syncthreads();   // previous Phase B finished reading As
    {                  // SiLU(gate)*up in-registers -> bf16 activations in LDS
      const int col = wv * 16 + l15;
#pragma unroll
      for (int j = 0; j < 8; ++j) {
        As[(lhalf + j) * ASTR + col] = (bf16_t)(fast_silu(cg[j]) * cu[j]);
      }
    }
    __syncthreads();

    // -------- Phase B: down-projection, K = IC --------
    v16bf a2[4];
#pragma unroll
    for (int kt = 0; kt < 4; ++kt) a2[kt] = ld_a(As, kt * 32, ASTR, lane);

#pragma unroll
    for (int nt = 0; nt < 8; ++nt) {
      const int hrow = wv * 128 + nt * 16 + l15;   // row of d weight (H dim)
      const bf16_t* dRow = dp + (size_t)hrow * I_DIM + ic + lhalf;
#pragma unroll
      for (int kt = 0; kt < 4; ++kt) {
        v16bf bd = ld_frag(dRow + kt * 32);
        acc[nt] = wmma_bf16(a2[kt], bd, acc[nt]);
      }
    }
  }

  // -------- epilogue: C layout N=lane&15, M=vgpr + 8*(lane>=16) ----------
#pragma unroll
  for (int nt = 0; nt < 8; ++nt) {
    const int col = wv * 128 + nt * 16 + l15;
#pragma unroll
    for (int j = 0; j < 8; ++j) {
      const int r = lhalf + j;
      float v = acc[nt][j] * coef_s[r];
      float* o = out + (size_t)tok_s[r] * H_DIM + col;
      if (expert >= 0) *o += v;   // routed passes run sequentially: race-free
      else             *o = v;    // shared pass writes every element first
    }
  }
}

// ---- Host driver --------------------------------------------------------

extern "C" void kernel_launch(void* const* d_in, const int* in_sizes, int n_in,
                              void* d_out, int out_size, void* d_ws, size_t ws_size,
                              hipStream_t stream) {
  (void)in_sizes; (void)n_in; (void)out_size; (void)ws_size;
  const float* x  = (const float*)d_in[0];
  const float* sg = (const float*)d_in[1];
  const float* su = (const float*)d_in[2];
  const float* sd = (const float*)d_in[3];
  const float* rg = (const float*)d_in[4];
  const float* ru = (const float*)d_in[5];
  const float* rd = (const float*)d_in[6];
  const float* wr = (const float*)d_in[7];
  const float* rb = (const float*)d_in[8];
  float* out = (float*)d_out;

  char* p = (char*)d_ws;
  auto take = [&](size_t bytes) -> char* {
    char* r = p; p += (bytes + 255) & ~(size_t)255; return r;
  };
  bf16_t* xb  = (bf16_t*)take((size_t)NTOK * H_DIM * 2);
  bf16_t* sgb = (bf16_t*)take((size_t)I_DIM * H_DIM * 2);
  bf16_t* sub = (bf16_t*)take((size_t)I_DIM * H_DIM * 2);
  bf16_t* sdb = (bf16_t*)take((size_t)H_DIM * I_DIM * 2);
  bf16_t* rgb = (bf16_t*)take((size_t)E_R * I_DIM * H_DIM * 2);
  bf16_t* rub = (bf16_t*)take((size_t)E_R * I_DIM * H_DIM * 2);
  bf16_t* rdb = (bf16_t*)take((size_t)E_R * H_DIM * I_DIM * 2);
  int*    i0  = (int*)take((size_t)NTOK * 4);
  int*    i1  = (int*)take((size_t)NTOK * 4);
  float*  w0  = (float*)take((size_t)NTOK * 4);
  float*  w1  = (float*)take((size_t)NTOK * 4);
  int*    ltok   = (int*)take((size_t)E_R * NTOK * 4);
  float*  lwv    = (float*)take((size_t)E_R * NTOK * 4);
  int*    counts = (int*)take(256);

  auto cvt = [&](const float* src, bf16_t* dst, size_t n) {
    int blocks = (int)((n / 4 + 255) / 256);
    cvt_kernel<<<blocks, 256, 0, stream>>>(src, dst, (int)n);
  };
  cvt(x,  xb,  (size_t)NTOK * H_DIM);
  cvt(sg, sgb, (size_t)I_DIM * H_DIM);
  cvt(su, sub, (size_t)I_DIM * H_DIM);
  cvt(sd, sdb, (size_t)H_DIM * I_DIM);
  cvt(rg, rgb, (size_t)E_R * I_DIM * H_DIM);
  cvt(ru, rub, (size_t)E_R * I_DIM * H_DIM);
  cvt(rd, rdb, (size_t)E_R * H_DIM * I_DIM);

  router_kernel<<<NTOK / 256, 256, 0, stream>>>(x, wr, rb, i0, i1, w0, w1);
  build_lists_kernel<<<1, 32, 0, stream>>>(i0, i1, w0, w1, ltok, lwv, counts);

  // Shared expert: writes every output element (initializes d_out).
  moe_ffn_kernel<<<dim3(NTOK / TM), 256, 0, stream>>>(
      xb, sgb, sub, sdb, ltok, lwv, counts, out, -1);
  // Routed experts: sequential launches -> deterministic, atomic-free +=.
  for (int e = 0; e < E_R; ++e)
    moe_ffn_kernel<<<dim3(NTOK / TM), 256, 0, stream>>>(
        xb, rgb, rub, rdb, ltok, lwv, counts, out, e);
}
